// MarketTwinLSTM_36636071035506
// MI455X (gfx1250) — matrix operand order repaired and checked by reading, not compiled, not run on hardware
//
#include <hip/hip_runtime.h>
#include <hip/hip_bf16.h>

#define B_ 256
#define S_ 2048
#define D_ 16
#define H_ 128
#define G_ 512   // 4*H

typedef __attribute__((ext_vector_type(16))) __bf16 v16bf;
typedef __attribute__((ext_vector_type(8)))  float  v8f;

union FragB { v16bf v; unsigned short s[16]; uint4 q[2]; };
union FragC { v8f v; float f[8]; };

__device__ inline unsigned short f2bf(float f) {
    unsigned int u = __float_as_uint(f);
    u += 0x7FFFu + ((u >> 16) & 1u);      // round-to-nearest-even
    return (unsigned short)(u >> 16);
}

__device__ inline float sigmoidf_(float z) { return 1.0f / (1.0f + __expf(-z)); }
__device__ inline float softplusf_(float z) {
    return fmaxf(z, 0.0f) + log1pf(__expf(-fabsf(z)));
}

// ---- A fragment (16x32 bf16, MxK): lane<16 holds K {0..7,16..23}, lane>=16 {8..15,24..31}
__device__ inline v16bf load_afrag_lds(const unsigned short* lds, int ld, int chunk, int lane) {
    int m  = lane & 15;
    int k0 = chunk * 32 + ((lane & 16) ? 8 : 0);
    FragB fb;
    fb.q[0] = *(const uint4*)(lds + m * ld + k0);        // K = k0 .. k0+7
    fb.q[1] = *(const uint4*)(lds + m * ld + k0 + 16);   // K = k0+16 .. k0+23
    return fb.v;
}

// ---- B fragment (32x16 bf16, KxN): lane<16 holds K {0..7,16..23}, lane>=16 {8..15,24..31}
__device__ inline v16bf load_bfrag_lds(const unsigned short* lds, int ld, int chunk, int lane) {
    int n  = lane & 15;
    int k0 = chunk * 32 + ((lane & 16) ? 8 : 0);
    FragB fb;
    fb.q[0] = *(const uint4*)(lds + n * ld + k0);
    fb.q[1] = *(const uint4*)(lds + n * ld + k0 + 16);
    return fb.v;
}

// ---- B fragment built from global f32 weights: B[k][n] = W[n][k]; zero-pad k >= kvalid
__device__ inline v16bf load_wfrag(const float* __restrict__ W, int ld, int kvalid,
                                   int n, int kbase, int lane) {
    FragB fb;
    int k0 = kbase + ((lane & 16) ? 8 : 0);
    const float* row = W + (size_t)n * ld;
#pragma unroll
    for (int e = 0; e < 16; ++e) {
        int k = k0 + (e < 8 ? e : e + 8);
        fb.s[e] = (k < kvalid) ? f2bf(row[k]) : (unsigned short)0;
    }
    return fb.v;
}

// =====================================================================
// One LSTM layer. 16 workgroups; each owns 16 batch rows for all 2048
// steps. 512 threads = 16 wave32 waves; wave w owns gate N-tiles 2w,2w+1.
// LAYER==0: input = x (B,S,16) f32.  LAYER==1: input = hbuf bf16 (in-place h1->h2).
// =====================================================================
template <int LAYER>
__global__ __launch_bounds__(512)
void lstm_layer_kernel(const float* __restrict__ x,
                       unsigned short* __restrict__ hbuf,          // (B,S,H) bf16
                       const float* __restrict__ W_ih,             // (4H, Din)
                       const float* __restrict__ W_hh,             // (4H, H)
                       const float* __restrict__ b_ih,
                       const float* __restrict__ b_hh,
                       float* __restrict__ out_hh,                 // (2,B,H)
                       float* __restrict__ out_hc) {               // (2,B,H)
    constexpr int IN_CHUNKS = (LAYER == 0) ? 1 : 4;
    constexpr int IN_KVALID = (LAYER == 0) ? 16 : 128;
    constexpr int IN_LD     = (LAYER == 0) ? 32 : 128;
    constexpr int IN_W_LD   = (LAYER == 0) ? D_ : H_;

    __shared__ __align__(16) float          lds_g[16 * G_];     // gates, 32 KB
    __shared__ __align__(16) float          lds_c[16 * H_];     // cell state f32, 8 KB
    __shared__ __align__(16) unsigned short lds_h[16 * H_];     // h bf16, 4 KB
    __shared__ __align__(16) unsigned short lds_in[16 * 128];   // staged input tile

    const int tid  = threadIdx.x;
    const int lane = tid & 31;
    const int w    = tid >> 5;          // wave 0..15
    const int col  = lane & 15;
    const int hi8  = (lane & 16) ? 8 : 0;
    const int b0   = blockIdx.x * 16;

    // --- weights -> register-resident bf16 fragments (one-time) ---
    v16bf whh[2][4];
    v16bf wih[2][IN_CHUNKS];
    float bias[2];
#pragma unroll
    for (int i = 0; i < 2; ++i) {
        int n   = (w * 2 + i) * 16 + col;
        bias[i] = b_ih[n] + b_hh[n];
#pragma unroll
        for (int c = 0; c < 4; ++c)
            whh[i][c] = load_wfrag(W_hh, H_, H_, n, c * 32, lane);
#pragma unroll
        for (int c = 0; c < IN_CHUNKS; ++c)
            wih[i][c] = load_wfrag(W_ih, IN_W_LD, IN_KVALID, n, c * 32, lane);
    }

    // --- zero initial state ---
    for (int i = tid; i < 16 * H_; i += 512) { lds_c[i] = 0.0f; lds_h[i] = 0; }

    for (int t = 0; t < S_; ++t) {
        __syncthreads();   // prev elementwise done; lds_in free
        // ---- stage input tile into LDS ----
        if (LAYER == 0) {
            int m = tid >> 5, k = tid & 31;          // 512 == 16*32
            float f = (k < D_) ? x[(size_t)(b0 + m) * (S_ * D_) + t * D_ + k] : 0.0f;
            lds_in[m * 32 + k] = f2bf(f);
        } else {
            int i4 = tid * 4;
            int m = i4 >> 7, k = i4 & 127;
            const unsigned short* src = hbuf + ((size_t)(b0 + m) * S_ + t) * H_ + k;
            *(uint2*)(lds_in + m * 128 + k) = *(const uint2*)src;
            if (t + 1 < S_)
                __builtin_prefetch(hbuf + ((size_t)(b0 + m) * S_ + (t + 1)) * H_ + k, 0, 0);
        }
        __syncthreads();

        // ---- gates = h@W_hh^T + in@W_ih^T + bias : bf16 WMMA, f32 accumulate ----
        FragC acc[2];
#pragma unroll
        for (int i = 0; i < 2; ++i)
#pragma unroll
            for (int e = 0; e < 8; ++e) acc[i].f[e] = bias[i];

#pragma unroll
        for (int c = 0; c < 4; ++c) {
            v16bf a = load_afrag_lds(lds_h, H_, c, lane);
            acc[0].v = __builtin_amdgcn_wmma_f32_16x16x32_bf16(false, a, false, whh[0][c],
                                                               (short)0, acc[0].v, false, false);
            acc[1].v = __builtin_amdgcn_wmma_f32_16x16x32_bf16(false, a, false, whh[1][c],
                                                               (short)0, acc[1].v, false, false);
        }
#pragma unroll
        for (int c = 0; c < IN_CHUNKS; ++c) {
            v16bf a = load_afrag_lds(lds_in, IN_LD, c, lane);
            acc[0].v = __builtin_amdgcn_wmma_f32_16x16x32_bf16(false, a, false, wih[0][c],
                                                               (short)0, acc[0].v, false, false);
            acc[1].v = __builtin_amdgcn_wmma_f32_16x16x32_bf16(false, a, false, wih[1][c],
                                                               (short)0, acc[1].v, false, false);
        }

        // ---- scatter gate tiles to LDS (C layout: M = e + 8*(lane>=16), N = lane&15) ----
#pragma unroll
        for (int i = 0; i < 2; ++i) {
            int n = (w * 2 + i) * 16 + col;
#pragma unroll
            for (int e = 0; e < 8; ++e)
                lds_g[(e + hi8) * G_ + n] = acc[i].f[e];
        }
        __syncthreads();

        // ---- LSTM cell elementwise: 4 hidden units per thread ----
        {
            int base = tid * 4;
            int m = base >> 7, j0 = base & 127;
            unsigned short hb[4];
#pragma unroll
            for (int u = 0; u < 4; ++u) {
                int j = j0 + u;
                float zi = lds_g[m * G_ + j];
                float zf = lds_g[m * G_ + 128 + j];
                float zg = lds_g[m * G_ + 256 + j];
                float zo = lds_g[m * G_ + 384 + j];
                float iv = sigmoidf_(zi);
                float fv = sigmoidf_(zf);
                float gv = tanhf(zg);
                float ov = sigmoidf_(zo);
                float cn = fv * lds_c[m * H_ + j] + iv * gv;
                float hn = ov * tanhf(cn);
                lds_c[m * H_ + j] = cn;
                hb[u] = f2bf(hn);
                lds_h[m * H_ + j] = hb[u];
                if (t == S_ - 1) {
                    out_hh[(size_t)LAYER * (B_ * H_) + (size_t)(b0 + m) * H_ + j] = hn;
                    out_hc[(size_t)LAYER * (B_ * H_) + (size_t)(b0 + m) * H_ + j] = cn;
                }
            }
            uint2 pk;
            pk.x = (unsigned)hb[0] | ((unsigned)hb[1] << 16);
            pk.y = (unsigned)hb[2] | ((unsigned)hb[3] << 16);
            *(uint2*)(hbuf + ((size_t)(b0 + m) * S_ + t) * H_ + j0) = pk;
        }
    }
}

// =====================================================================
// Heads: per row r of h2 (B*S rows, 128 wide):
//   ret = Wr2 @ leaky(Wr1 @ h + br1) + br2 ; vol = softplus(same with Wv*)
// Wr1|Wv1 staged as one 128x128 bf16 LDS block; 64->1 folded in via per-lane
// scale + 16-lane butterfly reduction.
// =====================================================================
__global__ __launch_bounds__(256)
void heads_kernel(const unsigned short* __restrict__ hbuf,
                  const float* __restrict__ Wr1, const float* __restrict__ br1,
                  const float* __restrict__ Wr2, const float* __restrict__ br2,
                  const float* __restrict__ Wv1, const float* __restrict__ bv1,
                  const float* __restrict__ Wv2, const float* __restrict__ bv2,
                  float* __restrict__ out_ret, float* __restrict__ out_vol) {
    __shared__ __align__(16) unsigned short ldsW[128 * 128];   // 32 KB

    const int tid  = threadIdx.x;
    const int lane = tid & 31;
    const int wid  = tid >> 5;
    const int col  = lane & 15;
    const int hi8  = (lane & 16) ? 8 : 0;

    for (int i = tid; i < 128 * 128; i += 256) {
        int n = i >> 7, k = i & 127;
        float f = (n < 64) ? Wr1[n * 128 + k] : Wv1[(n - 64) * 128 + k];
        ldsW[i] = f2bf(f);
    }
    float w2v[8], bias1[8];
#pragma unroll
    for (int nt = 0; nt < 8; ++nt) {
        int n = nt * 16 + col;
        if (n < 64) { w2v[nt] = Wr2[n];      bias1[nt] = br1[n]; }
        else        { w2v[nt] = Wv2[n - 64]; bias1[nt] = bv1[n - 64]; }
    }
    const float br2s = br2[0], bv2s = bv2[0];
    __syncthreads();

    const int ntiles = (B_ * S_) / 16;   // 32768 M-tiles
    for (int rt = blockIdx.x * 8 + wid; rt < ntiles; rt += gridDim.x * 8) {
        size_t r0 = (size_t)rt * 16;
        const unsigned short* hrow = hbuf + (r0 + (size_t)col) * H_;
        v16bf a[4];
#pragma unroll
        for (int c = 0; c < 4; ++c) {
            FragB fb;
            int k0 = c * 32 + hi8;
            fb.q[0] = *(const uint4*)(hrow + k0);
            fb.q[1] = *(const uint4*)(hrow + k0 + 16);
            a[c] = fb.v;
        }
        float pr[8] = {0,0,0,0,0,0,0,0}, pv[8] = {0,0,0,0,0,0,0,0};
#pragma unroll
        for (int nt = 0; nt < 8; ++nt) {
            FragC acc;
#pragma unroll
            for (int e = 0; e < 8; ++e) acc.f[e] = bias1[nt];
#pragma unroll
            for (int c = 0; c < 4; ++c) {
                v16bf bfrag = load_bfrag_lds(ldsW + nt * 16 * 128, 128, c, lane);
                acc.v = __builtin_amdgcn_wmma_f32_16x16x32_bf16(false, a[c], false, bfrag,
                                                                (short)0, acc.v, false, false);
            }
#pragma unroll
            for (int e = 0; e < 8; ++e) {
                float z = acc.f[e];
                z = (z > 0.0f) ? z : 0.01f * z;          // leaky_relu(0.01)
                if (nt < 4) pr[e] += z * w2v[nt]; else pv[e] += z * w2v[nt];
            }
        }
#pragma unroll
        for (int e = 0; e < 8; ++e) {
#pragma unroll
            for (int off = 8; off >= 1; off >>= 1) {     // reduce within 16-lane halves
                pr[e] += __shfl_xor(pr[e], off, 32);
                pv[e] += __shfl_xor(pv[e], off, 32);
            }
        }
        if (col == 0) {
#pragma unroll
            for (int e = 0; e < 8; ++e) {
                size_t row = r0 + (size_t)(hi8 + e);
                out_ret[row] = pr[e] + br2s;
                out_vol[row] = softplusf_(pv[e] + bv2s);
            }
        }
    }
}

extern "C" void kernel_launch(void* const* d_in, const int* in_sizes, int n_in,
                              void* d_out, int out_size, void* d_ws, size_t ws_size,
                              hipStream_t stream) {
    const float* x     = (const float*)d_in[0];
    const float* W_ih0 = (const float*)d_in[1];
    const float* W_hh0 = (const float*)d_in[2];
    const float* b_ih0 = (const float*)d_in[3];
    const float* b_hh0 = (const float*)d_in[4];
    const float* W_ih1 = (const float*)d_in[5];
    const float* W_hh1 = (const float*)d_in[6];
    const float* b_ih1 = (const float*)d_in[7];
    const float* b_hh1 = (const float*)d_in[8];
    const float* Wr1   = (const float*)d_in[9];
    const float* br1   = (const float*)d_in[10];
    const float* Wr2   = (const float*)d_in[11];
    const float* br2   = (const float*)d_in[12];
    const float* Wv1   = (const float*)d_in[13];
    const float* bv1   = (const float*)d_in[14];
    const float* Wv2   = (const float*)d_in[15];
    const float* bv2   = (const float*)d_in[16];

    unsigned short* hbuf = (unsigned short*)d_ws;   // (B,S,H) bf16: 128 MB

    float* out      = (float*)d_out;
    float* out_ret  = out;                                    // (B,S,1)
    float* out_vol  = out + (size_t)B_ * S_;                  // (B,S,1)
    float* out_hh   = out + (size_t)2 * B_ * S_;              // (2,B,H)
    float* out_hc   = out_hh + (size_t)2 * B_ * H_;           // (2,B,H)

    lstm_layer_kernel<0><<<B_ / 16, 512, 0, stream>>>(x, hbuf, W_ih0, W_hh0, b_ih0, b_hh0,
                                                      out_hh, out_hc);
    lstm_layer_kernel<1><<<B_ / 16, 512, 0, stream>>>(x, hbuf, W_ih1, W_hh1, b_ih1, b_hh1,
                                                      out_hh, out_hc);
    heads_kernel<<<512, 256, 0, stream>>>(hbuf, Wr1, br1, Wr2, br2, Wv1, bv1, Wv2, bv2,
                                          out_ret, out_vol);
}
